// MambaSSM_35751307771914
// MI455X (gfx1250) — compile-verified, matrix operand order checked by reading, not run on hardware
//
#include <hip/hip_runtime.h>
#include <cstdint>

// ---------------------------------------------------------------------------
// Mamba selective scan for MI455X (gfx1250, wave32).
//   Phase 1: fused projection GEMM  [4096 x 96]  = x_bf16 @ [W_B;W_C;W_dt_in]^T   (WMMA bf16)
//   Phase 2: dt GEMM + softplus     [4096 x 2048]= dt_low_bf16 @ W_dt_out^T + b   (WMMA bf16)
//   Phase 3: chunked linear scan (16 chunks of 128 steps): local pass -> prefix
//            fix-up -> replay pass emitting y.  Linearity of x_s = a*x + b makes
//            the chunk decomposition exact.
// ---------------------------------------------------------------------------

#define H       2048
#define SEQ     2048
#define BATCH   2
#define NSTATE  16
#define RANK    64
#define NC      16            // sequence chunks
#define LCH     (SEQ / NC)    // 128 steps per chunk
#define POUT    96            // fused projection width: 16 (B) + 16 (C) + 64 (dt_low)

typedef __attribute__((ext_vector_type(16))) __bf16 v16bf;
typedef __attribute__((ext_vector_type(8)))  float  v8f;

struct BF16Frag { uint4 lo, hi; };   // 32 bytes == v16bf

static __device__ __forceinline__ unsigned short f2bf(float f) {
    union { float f; unsigned int u; } v; v.f = f;
    unsigned int r = v.u + 0x7fffu + ((v.u >> 16) & 1u);   // round-to-nearest-even
    return (unsigned short)(r >> 16);
}

// ---- fp32 -> bf16 bulk convert (vectorized: float4 in, ushort4 out) --------
__global__ void k_f32_to_bf16(const float* __restrict__ src,
                              unsigned short* __restrict__ dst, int n4) {
    int i = blockIdx.x * blockDim.x + threadIdx.x;
    if (i >= n4) return;
    float4 f = ((const float4*)src)[i];
    ushort4 o;
    o.x = f2bf(f.x); o.y = f2bf(f.y); o.z = f2bf(f.z); o.w = f2bf(f.w);
    ((ushort4*)dst)[i] = o;
}

// ---- build concatenated projection weights [96][H] in bf16 -----------------
__global__ void k_build_wcat(const float* __restrict__ WB, const float* __restrict__ WC,
                             const float* __restrict__ Wdtin, unsigned short* __restrict__ wcat) {
    int i = blockIdx.x * blockDim.x + threadIdx.x;         // over 96*H
    if (i >= POUT * H) return;
    int o = i / H, k = i % H;
    const float* src = (o < 16) ? (WB + (size_t)o * H)
                     : (o < 32) ? (WC + (size_t)(o - 16) * H)
                                : (Wdtin + (size_t)(o - 32) * H);
    wcat[i] = f2bf(src[k]);
}

// ---- GEMM 1: P[m][0..95] = x[m][:] . wcat[o][:]  (M=4096, N=96, K=2048) ----
// One 16x16 output tile per wave, 6 waves/block cover the 6 N-tiles of one
// M-tile.  K-loop of 32 with v_wmma_f32_16x16x32_bf16, fragments fetched as
// two global_load_b128 per operand per step.
__global__ void __launch_bounds__(192)
k_gemm_proj(const unsigned short* __restrict__ inb, const unsigned short* __restrict__ wcat,
            float* __restrict__ P, unsigned short* __restrict__ Pdtb) {
    const int lane  = threadIdx.x;          // 0..31 (wave32: blockDim.x == 32)
    const int ntile = threadIdx.y;          // 0..5
    const int mtile = blockIdx.x;           // 0..255
    const int l15   = lane & 15;
    const int kbase = (lane >> 4) * 8;      // ISA 16-bit A/B fragment layout
    const unsigned short* arow = inb  + (size_t)(mtile * 16 + l15) * H + kbase;
    const unsigned short* brow = wcat + (size_t)(ntile * 16 + l15) * H + kbase;
    v8f acc = {};
    #pragma unroll 4
    for (int kk = 0; kk < H; kk += 32) {
        BF16Frag fa, fb;
        fa.lo = *(const uint4*)(arow + kk);
        fa.hi = *(const uint4*)(arow + kk + 16);
        fb.lo = *(const uint4*)(brow + kk);
        fb.hi = *(const uint4*)(brow + kk + 16);
        acc = __builtin_amdgcn_wmma_f32_16x16x32_bf16(
                  false, __builtin_bit_cast(v16bf, fa),
                  false, __builtin_bit_cast(v16bf, fb),
                  (short)0, acc, false, false);
    }
    const int ncol = ntile * 16 + l15;
    #pragma unroll
    for (int r = 0; r < 8; ++r) {           // C/D layout: VGPR r -> M = r + 8*(lane>=16)
        int m = mtile * 16 + r + (lane >> 4) * 8;
        float v = acc[r];
        P[(size_t)m * POUT + ncol] = v;
        if (ncol >= 32)                      // dt_low columns also shadowed in bf16
            Pdtb[(size_t)m * RANK + (ncol - 32)] = f2bf(v);
    }
}

// ---- GEMM 2: dt[m][h] = softplus(Pdt[m][:] . W_dt_out[h][:] + b[h]) --------
// M=4096, N=2048, K=64 -> 2 WMMA steps per tile; 8 waves/block.
__global__ void __launch_bounds__(256)
k_gemm_dt(const unsigned short* __restrict__ Pdtb, const unsigned short* __restrict__ Wdtb,
          const float* __restrict__ bias, float* __restrict__ dtf) {
    const int lane  = threadIdx.x;
    const int ntile = blockIdx.y * 8 + threadIdx.y;   // 0..127
    const int mtile = blockIdx.x;                     // 0..255
    const int l15   = lane & 15;
    const int kbase = (lane >> 4) * 8;
    const unsigned short* arow = Pdtb + (size_t)(mtile * 16 + l15) * RANK + kbase;
    const unsigned short* brow = Wdtb + (size_t)(ntile * 16 + l15) * RANK + kbase;
    v8f acc = {};
    #pragma unroll
    for (int kk = 0; kk < RANK; kk += 32) {
        BF16Frag fa, fb;
        fa.lo = *(const uint4*)(arow + kk);
        fa.hi = *(const uint4*)(arow + kk + 16);
        fb.lo = *(const uint4*)(brow + kk);
        fb.hi = *(const uint4*)(brow + kk + 16);
        acc = __builtin_amdgcn_wmma_f32_16x16x32_bf16(
                  false, __builtin_bit_cast(v16bf, fa),
                  false, __builtin_bit_cast(v16bf, fb),
                  (short)0, acc, false, false);
    }
    const int hcol = ntile * 16 + l15;
    const float bh = bias[hcol];
    #pragma unroll
    for (int r = 0; r < 8; ++r) {
        int m = mtile * 16 + r + (lane >> 4) * 8;
        float v = acc[r] + bh;
        v = (v > 20.0f) ? v : log1pf(__expf(v));      // softplus
        dtf[(size_t)m * H + hcol] = v;
    }
}

// ---- chunked scan: pass A (passC=0) runs each chunk from x=0 and records the
// end state + per-state decay product; pass C (passC=1) replays from the
// corrected initial state and emits y. ---------------------------------------
__global__ void __launch_bounds__(256)
k_scan_chunk(const float* __restrict__ input, const float* __restrict__ P,
             const float* __restrict__ dtf,   const float* __restrict__ A_log,
             const float* __restrict__ Dvec,
             const float* __restrict__ chunkInit,
             float* __restrict__ chunkState, float* __restrict__ chunkProd,
             float* __restrict__ out, int passC) {
    __shared__ float sBC[32];                // B row [0..15], C row [16..31]
    const int tid = threadIdx.x;
    const int hc  = blockIdx.x & 7;
    const int c   = (blockIdx.x >> 3) & (NC - 1);
    const int b   = blockIdx.x >> 7;         // / (8*NC)
    const int h   = hc * 256 + tid;

    float A[NSTATE];
    #pragma unroll
    for (int n = 0; n < NSTATE; ++n) A[n] = -__expf(A_log[h * NSTATE + n]);
    const float Dh = Dvec[h];

    float x[NSTATE], pa[NSTATE];
    #pragma unroll
    for (int n = 0; n < NSTATE; ++n) { x[n] = 0.0f; pa[n] = 1.0f; }
    if (passC) {
        #pragma unroll
        for (int n = 0; n < NSTATE; ++n)
            x[n] = chunkInit[(((size_t)b * NC + c) * NSTATE + n) * H + h];
    }

    for (int i = 0; i < LCH; ++i) {
        const int s = c * LCH + i;
        const size_t row = (size_t)b * SEQ + s;
        if (tid < 32) sBC[tid] = P[row * POUT + tid];
        __syncthreads();
        const float u  = input[row * H + h];
        const float dt = dtf[row * H + h];
        float y = Dh * u;
        #pragma unroll
        for (int n = 0; n < NSTATE; ++n) {
            const float a = __expf(A[n] * dt);        // a <= 1, no overflow
            x[n] = a * x[n] + (dt * sBC[n]) * u;
            if (passC) y += sBC[16 + n] * x[n];
            else       pa[n] *= a;
        }
        if (passC) out[row * H + h] = y;
        __syncthreads();
    }
    if (!passC) {
        #pragma unroll
        for (int n = 0; n < NSTATE; ++n) {
            const size_t idx = (((size_t)b * NC + c) * NSTATE + n) * H + h;
            chunkState[idx] = x[n];
            chunkProd[idx]  = pa[n];
        }
    }
}

// ---- pass B: 16-step serial prefix over chunks, per (b,h,n) ----------------
__global__ void __launch_bounds__(256)
k_chunk_prefix(const float* __restrict__ chunkState, const float* __restrict__ chunkProd,
               float* __restrict__ chunkInit) {
    const int id = blockIdx.x * blockDim.x + threadIdx.x;  // B*NSTATE*H = 65536
    if (id >= BATCH * NSTATE * H) return;
    const int h = id % H;
    const int n = (id / H) % NSTATE;
    const int b = id / (H * NSTATE);
    float carry = 0.0f;
    for (int c = 0; c < NC; ++c) {
        const size_t idx = (((size_t)b * NC + c) * NSTATE + n) * H + h;
        chunkInit[idx] = carry;
        carry = chunkProd[idx] * carry + chunkState[idx];
    }
}

extern "C" void kernel_launch(void* const* d_in, const int* in_sizes, int n_in,
                              void* d_out, int out_size, void* d_ws, size_t ws_size,
                              hipStream_t stream) {
    (void)in_sizes; (void)n_in; (void)out_size; (void)ws_size;
    const float* input    = (const float*)d_in[0];
    const float* W_dt_in  = (const float*)d_in[1];
    const float* W_dt_out = (const float*)d_in[2];
    const float* b_dt_out = (const float*)d_in[3];
    const float* W_B      = (const float*)d_in[4];
    const float* W_C      = (const float*)d_in[5];
    const float* Dvec     = (const float*)d_in[6];
    const float* A_log    = (const float*)d_in[7];
    float* out = (float*)d_out;

    const size_t M = (size_t)BATCH * SEQ;   // 4096
    char* ws = (char*)d_ws;
    size_t off = 0;
    auto wsalloc = [&](size_t bytes) {
        char* p = ws + off; off += (bytes + 255) & ~(size_t)255; return p;
    };
    // ~66 MB total workspace, all regions fully overwritten before first read.
    unsigned short* inb  = (unsigned short*)wsalloc(M * H * 2);                          // 16.8 MB
    unsigned short* wcat = (unsigned short*)wsalloc((size_t)POUT * H * 2);               //  0.4 MB
    unsigned short* Pdtb = (unsigned short*)wsalloc(M * RANK * 2);                       //  0.5 MB
    unsigned short* Wdtb = (unsigned short*)wsalloc((size_t)H * RANK * 2);               //  0.3 MB
    float* P       = (float*)wsalloc(M * POUT * 4);                                      //  1.6 MB
    float* dtf     = (float*)wsalloc(M * H * 4);                                         // 33.6 MB
    float* chState = (float*)wsalloc((size_t)BATCH * NC * NSTATE * H * 4);               //  4.2 MB
    float* chProd  = (float*)wsalloc((size_t)BATCH * NC * NSTATE * H * 4);               //  4.2 MB
    float* chInit  = (float*)wsalloc((size_t)BATCH * NC * NSTATE * H * 4);               //  4.2 MB

    // bf16 shadows
    {
        int n4 = (int)(M * H / 4);
        k_f32_to_bf16<<<(n4 + 255) / 256, 256, 0, stream>>>(input, inb, n4);
        int nw = (int)((size_t)H * RANK / 4);
        k_f32_to_bf16<<<(nw + 255) / 256, 256, 0, stream>>>(W_dt_out, Wdtb, nw);
        int nc = POUT * H;
        k_build_wcat<<<(nc + 255) / 256, 256, 0, stream>>>(W_B, W_C, W_dt_in, wcat);
    }
    // WMMA GEMMs
    k_gemm_proj<<<dim3((unsigned)(M / 16)), dim3(32, 6), 0, stream>>>(inb, wcat, P, Pdtb);
    k_gemm_dt  <<<dim3((unsigned)(M / 16), H / 128), dim3(32, 8), 0, stream>>>(Pdtb, Wdtb, b_dt_out, dtf);
    // chunked scan
    k_scan_chunk<<<BATCH * NC * 8, 256, 0, stream>>>(input, P, dtf, A_log, Dvec,
                                                     nullptr, chState, chProd, nullptr, 0);
    k_chunk_prefix<<<(BATCH * NSTATE * H + 255) / 256, 256, 0, stream>>>(chState, chProd, chInit);
    k_scan_chunk<<<BATCH * NC * 8, 256, 0, stream>>>(input, P, dtf, A_log, Dvec,
                                                     chInit, nullptr, nullptr, out, 1);
}
